// Fstar_5884105196204
// MI455X (gfx1250) — compile-verified, matrix operand order checked by reading, not run on hardware
//
#include <hip/hip_runtime.h>
#include <hip/hip_bf16.h>
#include <stdint.h>

#define NXX 96
#define NN  9216        // 96*96
#define BATCH 64

typedef __bf16 v16bf __attribute__((ext_vector_type(16)));
typedef float  v8f   __attribute__((ext_vector_type(8)));
typedef unsigned int u32x4 __attribute__((ext_vector_type(4)));
typedef int          i32x4 __attribute__((ext_vector_type(4)));
typedef int          i32x8 __attribute__((ext_vector_type(8)));

union BF16x16 { v16bf v; uint32_t u[8]; };
union F32x8   { v8f   v; float    f[8]; };

#if defined(__gfx1250__) && __has_builtin(__builtin_amdgcn_tensor_load_to_lds) && __has_builtin(__builtin_amdgcn_s_wait_tensorcnt)
#define USE_TDM 1
#else
#define USE_TDM 0
#endif

// f32 -> bf16 with round-to-nearest-even
__device__ __forceinline__ uint16_t f2bf(float x) {
  union { float f; uint32_t u; } c; c.f = x;
  uint32_t r = c.u + 0x7FFFu + ((c.u >> 16) & 1u);
  return (uint16_t)(r >> 16);
}
__device__ __forceinline__ uint32_t pack2bf(float lo, float hi) {
  return (uint32_t)f2bf(lo) | ((uint32_t)f2bf(hi) << 16);
}

// ---------------- workspace layout (bytes) ----------------
// AbfP  : 4t x 2par x 96j x 96dw  uint32  =  294912 B  (parity-doubled A, bf16 pairs)
// Bwd   : 4t x 192x x 96j         float   =  294912 B  (row-doubled post*W)
// polar : 64 x 9216               bf16    = 1179648 B
// partial: 4 x 64 x 9216          float   = 9437184 B
// img_bf: 64 x 2 x 9216           bf16    = 2359296 B
#define OFF_ABFP   0
#define OFF_BWD    294912
#define OFF_POLAR  589824
#define OFF_PART   1769472
#define OFF_IMG    11206656

struct PrepArgs {
  const float* pre[4];
  const float* K[4];    // inner kernels (matmul side)
  const float* post[4];
  const float* W[4];    // outer elementwise kernels
};

// Kernel 0a: fold pre into K (parity-doubled, j-major, bf16) and post into W (row-doubled f32)
__global__ void prep_kernel(PrepArgs a, uint32_t* __restrict__ AbfP,
                            float* __restrict__ Bwd) {
  int gid = blockIdx.x * blockDim.x + threadIdx.x;
  if (gid < 4 * 2 * NXX * NXX) {
    // layout: (((t*2+par)*96 + j)*96 + y), dword y covers elements x0=2y+par, x0+1
    int y   = gid % NXX;
    int row = gid / NXX;
    int j   = row % NXX;
    int tp  = row / NXX;
    int par = tp & 1;
    int t   = tp >> 1;
    int x0  = 2 * y + par;
    int k0  = x0 % NXX;
    int k1  = (x0 + 1) % NXX;
    AbfP[gid] = pack2bf(a.pre[t][k0] * a.K[t][k0 * NXX + j],
                        a.pre[t][k1] * a.K[t][k1 * NXX + j]);
  }
  int g2 = gid - 4 * 2 * NXX * NXX;
  if (g2 >= 0 && g2 < 4 * 192 * NXX) {
    int t   = g2 / (192 * NXX);
    int rem = g2 - t * 192 * NXX;
    int x   = rem / NXX;         // doubled row 0..191
    int j   = rem % NXX;
    int r   = x % NXX;
    Bwd[g2] = a.post[t][r] * a.W[t][r * NXX + j];
  }
}

// Kernel 0b: convert inputs (f32) to bf16 images so the TDM can DMA them
__global__ void img_kernel(const float* __restrict__ inputs,
                           uint32_t* __restrict__ img_bf) {
  int idx = blockIdx.x * blockDim.x + threadIdx.x;
  if (idx >= BATCH * 2 * NN / 2) return;
  float2 v = *(const float2*)(inputs + 2 * (size_t)idx);
  img_bf[idx] = pack2bf(v.x, v.y);
}

// Kernel 1: fused rotate + (Img @ A_rot) + weighted row-reduction -> polar (bf16)
__global__ __launch_bounds__(256)
void polar_kernel(const uint16_t* __restrict__ img_bf,
                  const uint32_t* __restrict__ AbfP,
                  const float* __restrict__ Bwd,
                  uint16_t* __restrict__ polar_bf) {
  __shared__ uint16_t sData[2][NN];   // R and I images, bf16 (36 KB)
  const int b     = blockIdx.x;
  const int ibase = blockIdx.y * 16;
  const int tid   = threadIdx.x;
  const uint16_t* src = img_bf + (size_t)b * 2 * NN;

#if USE_TDM
  if (tid < 32) {   // one TDM issue from wave 0; EXEC is ignored by tensor ops
    uint32_t ldsoff = (uint32_t)(uintptr_t)(__attribute__((address_space(3))) uint16_t*)&sData[0][0];
    uint64_t gaddr  = (uint64_t)(uintptr_t)src;
    u32x4 g0;
    g0.x = 1u;                                   // count=1 (valid user descriptor)
    g0.y = ldsoff;                               // lds_addr
    g0.z = (uint32_t)gaddr;                      // global_addr[31:0]
    g0.w = (uint32_t)((gaddr >> 32) & 0x01FFFFFFu) | (2u << 30);  // addr[56:32] | type=2
    i32x8 g1;
    g1[0] = 0x00030000;                          // workgroup_mask=0, data_size=3 (8B)
    g1[1] = (int)(4608u << 16);                  // tensor_dim0 = 4608 (low 16 bits)
    g1[2] = (int)(1u << 16);                     // tensor_dim0 hi=0, tensor_dim1 = 1
    g1[3] = (int)(4608u << 16);                  // tensor_dim1 hi=0, tile_dim0 = 4608
    g1[4] = 1;                                   // tile_dim1 = 1, tile_dim2 = 0
    g1[5] = 4608;                                // tensor_dim0_stride (low 32)
    g1[6] = 0;                                   // stride0 hi, stride1 lo
    g1[7] = 0;                                   // stride1 hi
    i32x4 z4 = {0, 0, 0, 0};
#if __clang_major__ >= 23
    i32x8 z8 = {0, 0, 0, 0, 0, 0, 0, 0};
    __builtin_amdgcn_tensor_load_to_lds(g0, g1, z4, z4, z8, 0);
#else
    __builtin_amdgcn_tensor_load_to_lds(g0, g1, z4, z4, 0);
#endif
    __builtin_amdgcn_s_wait_tensorcnt((short)0);
  }
  __syncthreads();
#else
  {
    const uint32_t* s32 = (const uint32_t*)src;
    uint32_t* d32 = (uint32_t*)&sData[0][0];
    for (int idx = tid; idx < NN; idx += 256)    // 2*NN u16 == NN dwords
      d32[idx] = s32[idx];
    __syncthreads();
  }
#endif

  const int wave = tid >> 5;
  const int lane = tid & 31;
  const int jl   = lane & 15;          // N-index within tile (also M-index for A frags)
  const int khi  = (lane >> 4) << 3;   // +8 K offset for upper half-wave

  for (int ii = 0; ii < 2; ++ii) {
    const int i   = ibase + wave * 2 + ii;   // rotation index
    const int par = i & 1;
    for (int jt = 0; jt < 6; ++jt) {
      const int jg = jt * 16 + jl;
      float colacc = 0.f;
      for (int rt = 0; rt < 6; ++rt) {
        F32x8 acc[4] = {};
        for (int kt = 0; kt < 3; ++kt) {
          BF16x16 fR, fI, fb[4];
          const int r = rt * 16 + jl;
#pragma unroll
          for (int p = 0; p < 8; ++p) {
            const int k = kt * 32 + ((p >= 4) ? 16 : 0) + ((p & 3) << 1) + khi;
            fR.u[p] = *(const uint32_t*)&sData[0][r * NXX + k];
            fI.u[p] = *(const uint32_t*)&sData[1][r * NXX + k];
            const int x = k - i + NXX;       // doubled-row index, in [1,191]
            const int y = x >> 1;            // aligned dword in parity copy
#pragma unroll
            for (int t = 0; t < 4; ++t)
              fb[t].u[p] = AbfP[(((t << 1) | par) * NXX + jg) * NXX + y];
          }
          acc[0].v = __builtin_amdgcn_wmma_f32_16x16x32_bf16(false, fR.v, false, fb[0].v, (short)0, acc[0].v, false, false);
          acc[1].v = __builtin_amdgcn_wmma_f32_16x16x32_bf16(false, fR.v, false, fb[1].v, (short)0, acc[1].v, false, false);
          acc[2].v = __builtin_amdgcn_wmma_f32_16x16x32_bf16(false, fI.v, false, fb[2].v, (short)0, acc[2].v, false, false);
          acc[3].v = __builtin_amdgcn_wmma_f32_16x16x32_bf16(false, fI.v, false, fb[3].v, (short)0, acc[3].v, false, false);
        }
        // weighted row-reduction via row-doubled Bwd: no mod/select needed
        const int xbase = rt * 16 + khi - i + NXX;   // in [1,191] for v=0..7
#pragma unroll
        for (int v = 0; v < 8; ++v) {
          const int off = (xbase + v) * NXX + jg;
          colacc += Bwd[off]                 * acc[0].f[v]
                  + Bwd[192 * NXX + off]     * acc[1].f[v]
                  + Bwd[2 * 192 * NXX + off] * acc[2].f[v]
                  + Bwd[3 * 192 * NXX + off] * acc[3].f[v];
        }
      }
      // add the two half-wave partial sums (rows 0-7 vs 8-15 of each tile)
      union { float f; int i; } cv; cv.f = colacc;
      cv.i = __builtin_amdgcn_ds_swizzle(cv.i, 0x401F);  // SWAPX16
      const float tot = colacc + cv.f;
      if (lane < 16)
        polar_bf[(size_t)b * NN + i * NXX + jg] = f2bf(tot);
    }
  }
}

// Kernel 2: out[b,m] = sum_n polar[b,n] * cart[m,n]; K split 4-way into partials
__global__ __launch_bounds__(256)
void cart_kernel(const float* __restrict__ cart,
                 const uint16_t* __restrict__ polar_bf,
                 float* __restrict__ partial) {
  const int tid  = threadIdx.x;
  const int W    = blockIdx.x * 8 + (tid >> 5);
  const int lane = tid & 31;
  const int nt   = W >> 2;     // 576 column tiles of 16 m-values
  const int kc   = W & 3;      // K chunk
  const int jl   = lane & 15;
  const int khi  = (lane >> 4) << 3;
  const int m    = nt * 16 + jl;
  const float* cartRow = cart + (size_t)m * NN;

  F32x8 acc[4] = {};
  const int ks0 = kc * 72, ks1 = ks0 + 72;
  for (int ks = ks0; ks < ks1; ++ks) {
    const int kg = ks * 32;
    BF16x16 bfrag;
#pragma unroll
    for (int p = 0; p < 8; ++p) {
      const int n0 = kg + ((p >= 4) ? 16 : 0) + ((p & 3) << 1) + khi;
      const float2 xy = *(const float2*)(cartRow + n0);
      bfrag.u[p] = pack2bf(xy.x, xy.y);
    }
    if (ks + 2 < ks1) __builtin_prefetch(cartRow + (ks + 2) * 32, 0, 1);
#pragma unroll
    for (int mt = 0; mt < 4; ++mt) {
      BF16x16 afrag;
      const uint16_t* prow = polar_bf + (size_t)(mt * 16 + jl) * NN;
#pragma unroll
      for (int p = 0; p < 8; ++p) {
        const int n0 = kg + ((p >= 4) ? 16 : 0) + ((p & 3) << 1) + khi;
        afrag.u[p] = *(const uint32_t*)(prow + n0);
      }
      acc[mt].v = __builtin_amdgcn_wmma_f32_16x16x32_bf16(false, afrag.v, false, bfrag.v, (short)0, acc[mt].v, false, false);
    }
  }
#pragma unroll
  for (int mt = 0; mt < 4; ++mt)
#pragma unroll
    for (int v = 0; v < 8; ++v) {
      const int bb = mt * 16 + khi + v;          // batch row
      partial[(size_t)(kc * BATCH + bb) * NN + m] = acc[mt].f[v];
    }
}

// Kernel 3: sum 4 K-partials -> f32 output
__global__ void reduce_kernel(const float* __restrict__ partial,
                              float* __restrict__ out) {
  const int idx = blockIdx.x * blockDim.x + threadIdx.x;
  if (idx >= BATCH * NN) return;
  out[idx] = partial[idx] + partial[BATCH * NN + idx]
           + partial[2 * BATCH * NN + idx] + partial[3 * BATCH * NN + idx];
}

extern "C" void kernel_launch(void* const* d_in, const int* in_sizes, int n_in,
                              void* d_out, int out_size, void* d_ws, size_t ws_size,
                              hipStream_t stream) {
  const float* inputs = (const float*)d_in[0];
  const float* cart   = (const float*)d_in[1];

  PrepArgs pa;
  pa.pre[0]  = (const float*)d_in[2];  pa.pre[1]  = (const float*)d_in[3];
  pa.pre[2]  = (const float*)d_in[4];  pa.pre[3]  = (const float*)d_in[5];
  pa.post[0] = (const float*)d_in[6];  pa.post[1] = (const float*)d_in[7];
  pa.post[2] = (const float*)d_in[8];  pa.post[3] = (const float*)d_in[9];
  // inner kernels (matmul side): cos2, sin2, sin3, cos4
  pa.K[0] = (const float*)d_in[12]; pa.K[1] = (const float*)d_in[13];
  pa.K[2] = (const float*)d_in[15]; pa.K[3] = (const float*)d_in[16];
  // outer elementwise kernels: cos1, sin1, cos3, sin4
  pa.W[0] = (const float*)d_in[10]; pa.W[1] = (const float*)d_in[11];
  pa.W[2] = (const float*)d_in[14]; pa.W[3] = (const float*)d_in[17];

  char* ws = (char*)d_ws;
  uint32_t* AbfP     = (uint32_t*)(ws + OFF_ABFP);
  float*    Bwd      = (float*)   (ws + OFF_BWD);
  uint16_t* polar_bf = (uint16_t*)(ws + OFF_POLAR);
  float*    partial  = (float*)   (ws + OFF_PART);
  uint32_t* img_bf   = (uint32_t*)(ws + OFF_IMG);

  // prep: 73728 AbfP dwords + 73728 Bwd floats = 147456 threads
  prep_kernel<<<(147456 + 255) / 256, 256, 0, stream>>>(pa, AbfP, Bwd);
  img_kernel<<<(BATCH * NN + 255) / 256, 256, 0, stream>>>(inputs, img_bf);
  polar_kernel<<<dim3(BATCH, 6), 256, 0, stream>>>((const uint16_t*)img_bf, AbfP, Bwd, polar_bf);
  cart_kernel<<<288, 256, 0, stream>>>(cart, polar_bf, partial);
  reduce_kernel<<<(BATCH * NN + 255) / 256, 256, 0, stream>>>(partial, (float*)d_out);
}
